// CfCNCPWrapper_78975858639370
// MI455X (gfx1250) — matrix operand order repaired and checked by reading, not compiled, not run on hardware
//
#include <hip/hip_runtime.h>
#include <hip/hip_bf16.h>
#include <math.h>

// ---------------------------------------------------------------------------
// CfC / NCP wired RNN for MI455X (gfx1250, wave32, WMMA 16x16x32 f16).
//
// Problem: scan over SEQ=256 steps; per step 3 layers, each with 4 matmuls
//   (128 x d) @ (d x n),  d/n = (333,269),(448,179),(243,64)
// Strategy:
//   * batch rows are independent across the recurrence -> 8 persistent
//     workgroups, each owns 16 batch rows (M=16 == WMMA M tile) and the full
//     256-step scan; hidden state lives in LDS as fp16.
//   * weights are mask-fused, converted to fp16, transposed & padded into
//     WMMA-B layout [Kpad][Npad] once per launch; streamed from L2 per step.
//   * v_wmma_f32_16x16x32_f16 with f32 accumulators; gating in f32 VALU.
// ---------------------------------------------------------------------------

typedef __attribute__((ext_vector_type(16))) _Float16 v16h;
typedef __attribute__((ext_vector_type(8)))  _Float16 v8h;
typedef __attribute__((ext_vector_type(8)))  float    v8f;

#define CFC_SEQ    256
#define CFC_BATCH  128
#define CFC_UNITS  512
#define CFC_OUT    64

// Layer geometry (derived from the reference):
//   l0: din=64,  n=269, d=333 -> Kpad=352 (11x32), Npad=272 (17x16)
//   l1: din=269, n=179, d=448 -> Kpad=448 (14x32), Npad=192 (12x16)
//   l2: din=179, n=64,  d=243 -> Kpad=256 ( 8x32), Npad=64  ( 4x16)
#define KP0 352
#define KP1 448
#define KP2 256

struct CfcParams {
    const float*    x;         // (BATCH, SEQ, 64) f32
    const float*    h0;        // (BATCH, 512) f32
    const _Float16* B[12];     // [layer*4 + {w1,w2,wa,wb}]: [Kpad][Npad] fp16
    const float*    bias[12];  // [layer*4 + {b1,b2,ba,bb}]
    float*          y;         // (BATCH, SEQ, 64) f32
    float*          hfin;      // (BATCH, 512) f32
};

// --- weight prep: out[k][n] = (k<d && n<nsize) ? w[n][k] * mask[n][k] : 0 ---
__global__ void cfc_prep_kernel(const float* __restrict__ w,
                                const float* __restrict__ mask,
                                _Float16* __restrict__ out,
                                int nsize, int d, int npad, int total)
{
    for (int i = blockIdx.x * blockDim.x + threadIdx.x; i < total;
         i += gridDim.x * blockDim.x) {
        const int k = i / npad;
        const int n = i % npad;
        float v = 0.0f;
        if (k < d && n < nsize) {
            v = w[(size_t)n * d + k];
            if (mask) v *= mask[(size_t)n * d + k];
        }
        out[i] = (_Float16)v;
    }
}

// --- one CfC layer step for a 16-row batch tile --------------------------
// xc : fp16 [16][KPAD] = [ inp (DIN cols) | h (NSIZE cols) | zero pad ]
// Computes: h_new = tanh(xc@W1m'+b1)*(1-t) + t*tanh(xc@W2m'+b2),
//           t = sigmoid(xc@Wa'+ba + xc@Wb'+bb)
// Writes h_new into xc's h-region and (if !LAST) into xcn's inp-region,
// else writes tanh(h_new) to global y.
template <int DIN, int NSIZE, int KPAD, int NPAD, int KT, int NT, int KPADN, bool LAST>
__device__ __forceinline__ void layer_step(
    _Float16* __restrict__ xc, _Float16* __restrict__ xcn,
    const _Float16* __restrict__ Bw1, const _Float16* __restrict__ Bw2,
    const _Float16* __restrict__ Bwa, const _Float16* __restrict__ Bwb,
    const float* __restrict__ b1, const float* __restrict__ b2,
    const float* __restrict__ ba, const float* __restrict__ bb,
    float* __restrict__ yout, int batch_base, int t)
{
    const int tid  = threadIdx.x;
    const int lane = tid & 31;
    const int wave = tid >> 5;
    const int half = lane >> 4;   // A: K-half select / C: M-half select
    const int msel = lane & 15;   // A: row M          / C: column N

    float outbuf[3][8];
    int   ntbuf[3];
    int   cnt = 0;

    // N-tile loop: condition is wave-uniform -> EXEC stays all-ones for WMMA.
    for (int nt = wave; nt < NT; nt += 8) {
        v8f acc1 = {}, acc2 = {}, acca = {}, accb = {};
        for (int kt = 0; kt < KT; ++kt) {
            // A fragment (16-bit A 16x32 layout): lane holds row M=msel,
            // halves 0..7 -> K = kt*32 + half*8 + {0..7},
            // halves 8..15 -> K = kt*32 + 16 + half*8 + {0..7}.
            const _Float16* ap = xc + msel * KPAD + kt * 32 + half * 8;
            const v8h alo = *(const v8h*)(ap);
            const v8h ahi = *(const v8h*)(ap + 16);
            v16h a;
#pragma unroll
            for (int i = 0; i < 8; ++i) { a[i] = alo[i]; a[i + 8] = ahi[i]; }

            // B fragment: lane -> N col (nt*16+msel), halves follow A's K order.
            v16h f1, f2, fa, fb;
#pragma unroll
            for (int i = 0; i < 16; ++i) {
                const int boff = (kt * 32 + half * 8 + (i & 7) + (i >> 3) * 16) * NPAD + nt * 16 + msel;
                f1[i] = Bw1[boff];
                f2[i] = Bw2[boff];
                fa[i] = Bwa[boff];
                fb[i] = Bwb[boff];
            }

            acc1 = __builtin_amdgcn_wmma_f32_16x16x32_f16(false, a, false, f1, (short)0, acc1, false, false);
            acc2 = __builtin_amdgcn_wmma_f32_16x16x32_f16(false, a, false, f2, (short)0, acc2, false, false);
            acca = __builtin_amdgcn_wmma_f32_16x16x32_f16(false, a, false, fa, (short)0, acca, false, false);
            accb = __builtin_amdgcn_wmma_f32_16x16x32_f16(false, a, false, fb, (short)0, accb, false, false);
        }

        // Gate math. C/D layout: lane -> N=msel, VGPR r -> M = half*8 + r.
        const int  nglob = nt * 16 + msel;
        const bool valid = (nglob < NSIZE);
        const float b1v = valid ? b1[nglob] : 0.0f;
        const float b2v = valid ? b2[nglob] : 0.0f;
        const float bav = valid ? ba[nglob] : 0.0f;
        const float bbv = valid ? bb[nglob] : 0.0f;
#pragma unroll
        for (int r = 0; r < 8; ++r) {
            const float ff1 = tanhf(acc1[r] + b1v);
            const float ff2 = tanhf(acc2[r] + b2v);
            const float g   = acca[r] + bav + accb[r] + bbv;
            const float s   = 1.0f / (1.0f + __expf(-g));
            outbuf[cnt][r]  = valid ? (ff1 * (1.0f - s) + s * ff2) : 0.0f;
        }
        ntbuf[cnt] = nt;
        ++cnt;
    }

    // All reads of xc done by every wave before anyone mutates it.
    __syncthreads();

    for (int i = 0; i < cnt; ++i) {
        const int nt    = ntbuf[i];
        const int nglob = nt * 16 + msel;
        if (nglob < NSIZE) {
#pragma unroll
            for (int r = 0; r < 8; ++r) {
                const int   m = half * 8 + r;
                const float o = outbuf[i][r];
                const _Float16 hv = (_Float16)o;
                xc[m * KPAD + DIN + nglob] = hv;           // new h_l (next step)
                if (!LAST) {
                    xcn[m * KPADN + nglob] = hv;           // next layer input
                } else {
                    yout[((size_t)(batch_base + m) * CFC_SEQ + t) * CFC_OUT + nglob] =
                        tanhf(o);
                }
            }
        }
    }
    __syncthreads();
}

// --- persistent scan kernel: one block per 16 batch rows ------------------
__global__ __launch_bounds__(256) void cfc_scan_kernel(CfcParams p)
{
    __shared__ __align__(16) _Float16 lds[16 * (KP0 + KP1 + KP2)];
    _Float16* xc0 = lds;
    _Float16* xc1 = xc0 + 16 * KP0;
    _Float16* xc2 = xc1 + 16 * KP1;

    const int batch_base = blockIdx.x * 16;
    const int tid = threadIdx.x;

    // Zero all xc buffers (padding must stay zero for WMMA over pad regions).
    for (int i = tid; i < 16 * (KP0 + KP1 + KP2); i += 256) lds[i] = (_Float16)0.0f;
    __syncthreads();

    // Seed hidden state from h0 (zeros in practice, but honor the input).
    for (int i = tid; i < 16 * 269; i += 256) {
        const int r = i / 269, c = i % 269;
        xc0[r * KP0 + 64 + c] = (_Float16)p.h0[(size_t)(batch_base + r) * CFC_UNITS + c];
    }
    for (int i = tid; i < 16 * 179; i += 256) {
        const int r = i / 179, c = i % 179;
        xc1[r * KP1 + 269 + c] = (_Float16)p.h0[(size_t)(batch_base + r) * CFC_UNITS + 269 + c];
    }
    for (int i = tid; i < 16 * 64; i += 256) {
        const int r = i >> 6, c = i & 63;
        xc2[r * KP2 + 179 + c] = (_Float16)p.h0[(size_t)(batch_base + r) * CFC_UNITS + 448 + c];
    }
    __syncthreads();

    for (int t = 0; t < CFC_SEQ; ++t) {
        // Stage this step's x tile (16 x 64) into layer-0 input columns.
        for (int i = tid; i < 16 * 64; i += 256) {
            const int r = i >> 6, c = i & 63;
            xc0[r * KP0 + c] =
                (_Float16)p.x[((size_t)(batch_base + r) * CFC_SEQ + t) * 64 + c];
        }
        __syncthreads();

        layer_step< 64, 269, KP0, 272, 11, 17, KP1, false>(
            xc0, xc1, p.B[0], p.B[1], p.B[2], p.B[3],
            p.bias[0], p.bias[1], p.bias[2], p.bias[3], nullptr, batch_base, t);
        layer_step<269, 179, KP1, 192, 14, 12, KP2, false>(
            xc1, xc2, p.B[4], p.B[5], p.B[6], p.B[7],
            p.bias[4], p.bias[5], p.bias[6], p.bias[7], nullptr, batch_base, t);
        layer_step<179,  64, KP2,  64,  8,  4,   1, true>(
            xc2, xc2, p.B[8], p.B[9], p.B[10], p.B[11],
            p.bias[8], p.bias[9], p.bias[10], p.bias[11], p.y, batch_base, t);
    }

    // Emit h_final = concat(h0_l, h1_l, h2_l) as f32.
    for (int i = tid; i < 16 * 269; i += 256) {
        const int r = i / 269, c = i % 269;
        p.hfin[(size_t)(batch_base + r) * CFC_UNITS + c] = (float)xc0[r * KP0 + 64 + c];
    }
    for (int i = tid; i < 16 * 179; i += 256) {
        const int r = i / 179, c = i % 179;
        p.hfin[(size_t)(batch_base + r) * CFC_UNITS + 269 + c] = (float)xc1[r * KP1 + 269 + c];
    }
    for (int i = tid; i < 16 * 64; i += 256) {
        const int r = i >> 6, c = i & 63;
        p.hfin[(size_t)(batch_base + r) * CFC_UNITS + 448 + c] = (float)xc2[r * KP2 + 179 + c];
    }
}

extern "C" void kernel_launch(void* const* d_in, const int* in_sizes, int n_in,
                              void* d_out, int out_size, void* d_ws, size_t ws_size,
                              hipStream_t stream)
{
    (void)in_sizes; (void)n_in; (void)out_size; (void)ws_size;

    // Input order: x, dt, h0, then per layer l: w1,w2,wa,wb,b1,b2,ba,bb,mask.
    static const int NS[3] = {269, 179, 64};
    static const int DD[3] = {333, 448, 243};
    static const int KP[3] = {KP0, KP1, KP2};
    static const int NP[3] = {272, 192, 64};

    CfcParams p;
    p.x    = (const float*)d_in[0];
    p.h0   = (const float*)d_in[2];
    p.y    = (float*)d_out;
    p.hfin = p.y + (size_t)CFC_BATCH * CFC_SEQ * CFC_OUT;

    _Float16* ws = (_Float16*)d_ws;
    size_t off = 0;
    for (int l = 0; l < 3; ++l) {
        const int base = 3 + l * 9;
        const float* mask = (const float*)d_in[base + 8];
        for (int j = 0; j < 4; ++j) {
            const int total = KP[l] * NP[l];
            _Float16* outp = ws + off;
            p.B[l * 4 + j]    = outp;
            p.bias[l * 4 + j] = (const float*)d_in[base + 4 + j];
            const float* w = (const float*)d_in[base + j];
            const float* m = (j < 2) ? mask : nullptr; // masks apply to w1,w2 only
            const int blocks = (total + 255) / 256;
            hipLaunchKernelGGL(cfc_prep_kernel, dim3(blocks), dim3(256), 0, stream,
                               w, m, outp, NS[l], DD[l], NP[l], total);
            off += (size_t)total;
        }
    }

    hipLaunchKernelGGL(cfc_scan_kernel, dim3(CFC_BATCH / 16), dim3(256), 0, stream, p);
}
